// GAT_74294344286957
// MI455X (gfx1250) — matrix operand order repaired and checked by reading, not compiled, not run on hardware
//
#include <hip/hip_runtime.h>
#include <hip/hip_bf16.h>
#include <cstdint>

typedef __attribute__((ext_vector_type(16))) _Float16 v16h;
typedef __attribute__((ext_vector_type(8)))  float    v8f;

#define N_NODES 50000
#define F_IN    165
#define KPAD    192      // 6 chunks of 32
#define NCHUNK  6
#define H1      4
#define D1      256      // H1 * 64
#define E0      800000
#define E_TOT   (E0 + N_NODES)
#define SLOPE   0.2f
#define ENC_NEG_INF 0x007FFFFFu   // order-preserving encoding of -inf

// ---------- order-preserving float <-> uint for atomic max ----------
__device__ __forceinline__ unsigned f2ord(float f) {
  unsigned u = __float_as_uint(f);
  return (u & 0x80000000u) ? ~u : (u | 0x80000000u);
}
__device__ __forceinline__ float ord2f(unsigned k) {
  return __uint_as_float((k & 0x80000000u) ? (k ^ 0x80000000u) : ~k);
}

__device__ __forceinline__ void load_edge(const int* __restrict__ ei, int e,
                                          int& s, int& d) {
  if (e < E0) { s = ei[e]; d = ei[E0 + e]; }
  else        { s = e - E0; d = s; }          // self loops
  s = min(max(s, 0), N_NODES - 1);
  d = min(max(d, 0), N_NODES - 1);
}

// ---------- init ----------
__global__ void fill_u32(unsigned* p, unsigned v, int n) {
  int i = blockIdx.x * blockDim.x + threadIdx.x;
  if (i < n) p[i] = v;
}

// ---------- W1 -> f16, zero-padded K 165 -> 192 ----------
__global__ void w1_to_f16(const float* __restrict__ W1, _Float16* __restrict__ W1h) {
  int i = blockIdx.x * blockDim.x + threadIdx.x;   // 0 .. KPAD*D1
  int k = i / D1, c = i % D1;
  float v = (k < F_IN) ? W1[k * D1 + c] : 0.f;
  W1h[i] = (_Float16)v;
}

// ---------- async 16B global -> LDS copy (ASYNCcnt-tracked) ----------
__device__ __forceinline__ void async_copy_b128(unsigned lds_byte_off,
                                                const void* gaddr) {
  asm volatile("global_load_async_to_lds_b128 %0, %1, off"
               :: "v"(lds_byte_off), "v"(gaddr) : "memory");
}

// stage one 32x256 f16 W-chunk (16 KB) into LDS: 512 thr x 2 x 16B
__device__ __forceinline__ void stage_w_async(const _Float16* __restrict__ W1h,
                                              int kc, unsigned lds_base, int tid) {
  const char* gbase = (const char*)(W1h + (size_t)kc * 32 * D1);
#pragma unroll
  for (int r = 0; r < 2; ++r) {
    unsigned byteoff = (unsigned)(tid + r * 512) * 16u;
    async_copy_b128(lds_base + byteoff, gbase + byteoff);
  }
}

// ---------- GEMM1: h1[50000,256] = x[50000,165] @ W1  (f16 WMMA, f32 acc) ----
// block = 512 threads = 16 waves; block computes a 16x256 tile; wave w owns
// 16 columns [16w,16w+16). K padded to 192, 6 chunks of 32.
// W chunks are double-buffered in LDS via async global->LDS copies.
__global__ __launch_bounds__(512)
void gemm1_wmma(const float* __restrict__ x, const _Float16* __restrict__ W1h,
                float* __restrict__ h1) {
  __shared__ __align__(16) _Float16 lx[16 * KPAD];      // A tile, 6 KB
  __shared__ __align__(16) _Float16 lw[2][32 * D1];     // 2 x 16 KB W chunks

  const int tid  = threadIdx.x;
  const int lane = tid & 31;
  const int wave = tid >> 5;
  const int row0 = blockIdx.x * 16;    // 50000/16 = 3125 exact tiles

  const unsigned lwoff0 = (unsigned)(uintptr_t)&lw[0][0]; // LDS byte offset
  const unsigned lwoff1 = (unsigned)(uintptr_t)&lw[1][0];

  // kick off async staging of W chunk 0, then convert A tile while it flies
  stage_w_async(W1h, 0, lwoff0, tid);
  for (int i = tid; i < 16 * KPAD; i += 512) {
    int r = i / KPAD, c = i % KPAD;
    lx[i] = (_Float16)((c < F_IN) ? x[(row0 + r) * F_IN + c] : 0.f);
  }

  v8f acc = {};
  const int mrow = lane & 15;                 // A: M = lane%16
  const int kba  = (lane >> 4) << 4;          // A: K half-offset 0/16
  const int ncol = (wave << 4) + (lane & 15); // global column this lane owns
  const int ksb  = (lane >> 4) << 4;          // B: K half-offset 0/16

  const unsigned* lx32 = (const unsigned*)lx; // A pairs are contiguous f16x2

#pragma unroll
  for (int kc = 0; kc < NCHUNK; ++kc) {
    const _Float16* lwc = lw[kc & 1];
    if (kc + 1 < NCHUNK) {
      stage_w_async(W1h, kc + 1, (kc & 1) ? lwoff0 : lwoff1, tid);
      asm volatile("s_wait_asynccnt 0x2" ::: "memory"); // chunk kc landed
    } else {
      asm volatile("s_wait_asynccnt 0x0" ::: "memory");
    }
    __syncthreads();   // all waves see chunk kc (and, first time, lx)

    union { v16h h; unsigned u[8]; } a;
    v16h b;
#pragma unroll
    for (int v = 0; v < 8; ++v) {
      // A 16x32 f16, contiguous k order used here (same as B): lanes0-15 K=0..15, lanes16-31 K=16..31
      int ka = kba + (v << 1);
      a.u[v] = lx32[(mrow * KPAD + kc * 32 + ka) >> 1];
      // B 32x16 f16, same contiguous k order as A: lanes0-15 K=0..15, lanes16-31 K=16..31, K-pairs per VGPR
      int kb = ksb + (v << 1);
      b[2 * v]     = lwc[kb * D1 + ncol];
      b[2 * v + 1] = lwc[(kb + 1) * D1 + ncol];
    }
    acc = __builtin_amdgcn_wmma_f32_16x16x32_f16(false, a.h, false, b,
                                                 (short)0, acc, false, false);
    __syncthreads();   // reads of chunk kc done before its buffer is reused
  }

  // C/D layout: VGPR r -> row r (lanes 0-15) or r+8 (lanes 16-31), col lane%16
  const int rbase = (lane >> 4) << 3;
#pragma unroll
  for (int r = 0; r < 8; ++r)
    h1[(size_t)(row0 + rbase + r) * D1 + ncol] = acc[r];
}

// ---------- per-node attention coefficients, layer 1 ----------
__global__ void attn1_coeff(const float* __restrict__ h1,
                            const float* __restrict__ att_src,
                            const float* __restrict__ att_dst,
                            float* __restrict__ as1, float* __restrict__ ad1) {
  __shared__ float ss[D1], sd[D1];
  int n = blockIdx.x, t = threadIdx.x;
  float hv = h1[(size_t)n * D1 + t];
  ss[t] = hv * att_src[t];
  sd[t] = hv * att_dst[t];
  __syncthreads();
  for (int off = 32; off >= 1; off >>= 1) {
    if ((t & 63) < off) { ss[t] += ss[t + off]; sd[t] += sd[t + off]; }
    __syncthreads();
  }
  if ((t & 63) == 0) {
    as1[n * H1 + (t >> 6)] = ss[t];
    ad1[n * H1 + (t >> 6)] = sd[t];
  }
}

// ---------- edge passes, layer 1 (H=4) ----------
__global__ void edge_max1(const int* __restrict__ ei, const float* __restrict__ as1,
                          const float* __restrict__ ad1, unsigned* __restrict__ m1) {
  int id = blockIdx.x * blockDim.x + threadIdx.x;
  if (id >= E_TOT * H1) return;
  int e = id >> 2, h = id & 3, s, d;
  load_edge(ei, e, s, d);
  float l = as1[s * H1 + h] + ad1[d * H1 + h];
  l = (l > 0.f) ? l : SLOPE * l;
  atomicMax(&m1[d * H1 + h], f2ord(l));
}

__global__ void edge_exp1(const int* __restrict__ ei, const float* __restrict__ as1,
                          const float* __restrict__ ad1, const unsigned* __restrict__ m1,
                          float* __restrict__ alpha, float* __restrict__ den) {
  int id = blockIdx.x * blockDim.x + threadIdx.x;
  if (id >= E_TOT * H1) return;
  int e = id >> 2, h = id & 3, s, d;
  load_edge(ei, e, s, d);
  float l = as1[s * H1 + h] + ad1[d * H1 + h];
  l = (l > 0.f) ? l : SLOPE * l;
  float a = __expf(l - ord2f(m1[d * H1 + h]));
  alpha[id] = a;
  atomicAdd(&den[d * H1 + h], a);
}

__global__ void edge_norm1(const int* __restrict__ ei, const float* __restrict__ den,
                           float* __restrict__ alpha) {
  int id = blockIdx.x * blockDim.x + threadIdx.x;
  if (id >= E_TOT * H1) return;
  int e = id >> 2, h = id & 3, s, d;
  load_edge(ei, e, s, d);
  alpha[id] = alpha[id] / den[d * H1 + h];
}

// one block (256 threads = 256 channels) per edge
__global__ __launch_bounds__(256)
void edge_scatter1(const int* __restrict__ ei, const float* __restrict__ h1,
                   const float* __restrict__ alpha, float* __restrict__ out1) {
  int e = blockIdx.x, c = threadIdx.x, s, d;
  load_edge(ei, e, s, d);
  __builtin_prefetch(&h1[(size_t)s * D1 + c], 0, 0);
  float w = alpha[e * H1 + (c >> 6)];
  atomicAdd(&out1[(size_t)d * D1 + c], h1[(size_t)s * D1 + c] * w);
}

// ---------- per-node: bias + ELU + (256->2) GEMV + layer-2 coefficients ----
__global__ void node2(const float* __restrict__ out1, const float* __restrict__ b1,
                      const float* __restrict__ W2, const float* __restrict__ att_src2,
                      const float* __restrict__ att_dst2, float* __restrict__ g,
                      float* __restrict__ as2, float* __restrict__ ad2) {
  __shared__ float s0[D1], s1[D1];
  int n = blockIdx.x, t = threadIdx.x;
  float v = out1[(size_t)n * D1 + t] + b1[t];
  float z = (v > 0.f) ? v : (__expf(v) - 1.f);   // ELU
  s0[t] = z * W2[t * 2 + 0];
  s1[t] = z * W2[t * 2 + 1];
  __syncthreads();
  for (int off = 128; off >= 1; off >>= 1) {
    if (t < off) { s0[t] += s0[t + off]; s1[t] += s1[t + off]; }
    __syncthreads();
  }
  if (t == 0) {
    float g0 = s0[0], g1 = s1[0];
    g[n * 2] = g0; g[n * 2 + 1] = g1;
    as2[n] = g0 * att_src2[0] + g1 * att_src2[1];
    ad2[n] = g0 * att_dst2[0] + g1 * att_dst2[1];
  }
}

// ---------- edge passes, layer 2 (H=1, C=2) ----------
__global__ void edge_max2(const int* __restrict__ ei, const float* __restrict__ as2,
                          const float* __restrict__ ad2, unsigned* __restrict__ m2) {
  int e = blockIdx.x * blockDim.x + threadIdx.x;
  if (e >= E_TOT) return;
  int s, d; load_edge(ei, e, s, d);
  float l = as2[s] + ad2[d];
  l = (l > 0.f) ? l : SLOPE * l;
  atomicMax(&m2[d], f2ord(l));
}

__global__ void edge_exp2(const int* __restrict__ ei, const float* __restrict__ as2,
                          const float* __restrict__ ad2, const unsigned* __restrict__ m2,
                          float* __restrict__ alpha2, float* __restrict__ den2) {
  int e = blockIdx.x * blockDim.x + threadIdx.x;
  if (e >= E_TOT) return;
  int s, d; load_edge(ei, e, s, d);
  float l = as2[s] + ad2[d];
  l = (l > 0.f) ? l : SLOPE * l;
  float a = __expf(l - ord2f(m2[d]));
  alpha2[e] = a;
  atomicAdd(&den2[d], a);
}

__global__ void edge_norm2(const int* __restrict__ ei, const float* __restrict__ den2,
                           float* __restrict__ alpha2) {
  int e = blockIdx.x * blockDim.x + threadIdx.x;
  if (e >= E_TOT) return;
  int s, d; load_edge(ei, e, s, d);
  alpha2[e] = alpha2[e] / den2[d];
}

__global__ void edge_scatter2(const int* __restrict__ ei, const float* __restrict__ g,
                              const float* __restrict__ alpha2, float* __restrict__ out2) {
  int id = blockIdx.x * blockDim.x + threadIdx.x;
  if (id >= E_TOT * 2) return;
  int e = id >> 1, c = id & 1, s, d;
  load_edge(ei, e, s, d);
  atomicAdd(&out2[d * 2 + c], g[s * 2 + c] * alpha2[e]);
}

// ---------- bias + log_softmax over 2 classes ----------
__global__ void logsm(const float* __restrict__ out2, const float* __restrict__ b2,
                      float* __restrict__ y) {
  int n = blockIdx.x * blockDim.x + threadIdx.x;
  if (n >= N_NODES) return;
  float y0 = out2[n * 2] + b2[0];
  float y1 = out2[n * 2 + 1] + b2[1];
  float m  = fmaxf(y0, y1);
  float lse = m + __logf(__expf(y0 - m) + __expf(y1 - m));
  y[n * 2]     = y0 - lse;
  y[n * 2 + 1] = y1 - lse;
}

extern "C" void kernel_launch(void* const* d_in, const int* in_sizes, int n_in,
                              void* d_out, int out_size, void* d_ws, size_t ws_size,
                              hipStream_t stream) {
  const float* x        = (const float*)d_in[0];
  const int*   ei       = (const int*)d_in[1];
  const float* W1       = (const float*)d_in[2];
  const float* att_src1 = (const float*)d_in[3];
  const float* att_dst1 = (const float*)d_in[4];
  const float* b1       = (const float*)d_in[5];
  const float* W2       = (const float*)d_in[6];
  const float* att_src2 = (const float*)d_in[7];
  const float* att_dst2 = (const float*)d_in[8];
  const float* b2       = (const float*)d_in[9];
  float* y = (float*)d_out;

  // ---- workspace carve-out (all offsets 256B aligned) ----
  char* ws = (char*)d_ws;
  size_t off = 0;
  auto carve = [&](size_t bytes) -> void* {
    void* p = ws + off;
    off = (off + bytes + 255) & ~(size_t)255;
    return p;
  };
  _Float16* W1h   = (_Float16*)carve((size_t)KPAD * D1 * 2);
  float*    h1    = (float*)carve((size_t)N_NODES * D1 * 4);
  float*    as1   = (float*)carve((size_t)N_NODES * H1 * 4);
  float*    ad1   = (float*)carve((size_t)N_NODES * H1 * 4);
  unsigned* m1    = (unsigned*)carve((size_t)N_NODES * H1 * 4);
  float*    den1  = (float*)carve((size_t)N_NODES * H1 * 4);
  float*    alpha = (float*)carve((size_t)E_TOT * H1 * 4);
  float*    out1  = (float*)carve((size_t)N_NODES * D1 * 4);
  float*    g     = (float*)carve((size_t)N_NODES * 2 * 4);
  float*    as2   = (float*)carve((size_t)N_NODES * 4);
  float*    ad2   = (float*)carve((size_t)N_NODES * 4);
  unsigned* m2    = (unsigned*)carve((size_t)N_NODES * 4);
  float*    den2  = (float*)carve((size_t)N_NODES * 4);
  float*    out2  = (float*)carve((size_t)N_NODES * 2 * 4);

  // ---- init accumulators (every call: deterministic under graph replay) ----
  hipMemsetAsync(den1, 0, (size_t)N_NODES * H1 * 4, stream);
  hipMemsetAsync(out1, 0, (size_t)N_NODES * D1 * 4, stream);
  hipMemsetAsync(den2, 0, (size_t)N_NODES * 4, stream);
  hipMemsetAsync(out2, 0, (size_t)N_NODES * 2 * 4, stream);
  fill_u32<<<(N_NODES * H1 + 255) / 256, 256, 0, stream>>>(m1, ENC_NEG_INF, N_NODES * H1);
  fill_u32<<<(N_NODES + 255) / 256, 256, 0, stream>>>(m2, ENC_NEG_INF, N_NODES);

  // ---- layer 1 ----
  w1_to_f16<<<(KPAD * D1 + 255) / 256, 256, 0, stream>>>(W1, W1h);
  gemm1_wmma<<<N_NODES / 16, 512, 0, stream>>>(x, W1h, h1);
  attn1_coeff<<<N_NODES, D1, 0, stream>>>(h1, att_src1, att_dst1, as1, ad1);
  edge_max1 <<<(E_TOT * H1 + 255) / 256, 256, 0, stream>>>(ei, as1, ad1, m1);
  edge_exp1 <<<(E_TOT * H1 + 255) / 256, 256, 0, stream>>>(ei, as1, ad1, m1, alpha, den1);
  edge_norm1<<<(E_TOT * H1 + 255) / 256, 256, 0, stream>>>(ei, den1, alpha);
  edge_scatter1<<<E_TOT, 256, 0, stream>>>(ei, h1, alpha, out1);

  // ---- layer 2 ----
  node2<<<N_NODES, D1, 0, stream>>>(out1, b1, W2, att_src2, att_dst2, g, as2, ad2);
  edge_max2 <<<(E_TOT + 255) / 256, 256, 0, stream>>>(ei, as2, ad2, m2);
  edge_exp2 <<<(E_TOT + 255) / 256, 256, 0, stream>>>(ei, as2, ad2, m2, alpha, den2);
  edge_norm2<<<(E_TOT + 255) / 256, 256, 0, stream>>>(ei, den2, alpha);
  edge_scatter2<<<(E_TOT * 2 + 255) / 256, 256, 0, stream>>>(ei, g, alpha, out2);

  logsm<<<(N_NODES + 255) / 256, 256, 0, stream>>>(out2, b2, y);
}
